// ICLCredibilityTransformer_4733053960677
// MI455X (gfx1250) — matrix operand order repaired and checked by reading, not compiled, not run on hardware
//
#include <hip/hip_runtime.h>
#include <hip/hip_bf16.h>

// ---------------------------------------------------------------------------
// CDNA5 (gfx1250) wave32 WMMA types
// ---------------------------------------------------------------------------
typedef __attribute__((ext_vector_type(16))) __bf16 v16bf;
typedef __attribute__((ext_vector_type(8)))  float  v8f;
typedef __attribute__((ext_vector_type(4)))  int    v4i;

template <bool B> struct BoolC { static constexpr bool value = B; };

// Async global->LDS (ASYNCcnt) path, if this toolchain exposes the gfx1250
// builtins; otherwise fall back to sync load + ds_store staging.
#if defined(__AMDGCN__) && __has_builtin(__builtin_amdgcn_global_load_async_to_lds_b128) && \
    __has_builtin(__builtin_amdgcn_s_wait_asynccnt)
#define HAVE_ASYNC_LDS 1
#else
#define HAVE_ASYNC_LDS 0
#endif

// param 1: generic (flat) non-const v4i* -> the global-memory source
// param 2: LDS (addrspace(3)) v4i*      -> the LDS destination
#define GPTR(x) ((v4i*)(x))
#define LPTR(x) ((__attribute__((address_space(3))) v4i*)(x))

__device__ inline unsigned short f2bf(float f) {
    unsigned int u = __float_as_uint(f);
    unsigned int r = (u + 0x7FFFu + ((u >> 16) & 1u)) >> 16;
    return (unsigned short)r;
}

__device__ inline v8f wmma_bf16(v16bf a, v16bf b, v8f c) {
    return __builtin_amdgcn_wmma_f32_16x16x32_bf16(
        /*neg_a=*/false, a, /*neg_b=*/false, b,
        /*c_mod=*/(short)0, c, /*reuse_a=*/false, /*reuse_b=*/false);
}

// A-matrix fragment (16x32 bf16, M=lane&15, K split per ISA layout):
// elements 0-7  <- row[group*8 + 0..7]
// elements 8-15 <- row[16 + group*8 + 0..7]
__device__ inline v16bf load_frag_a(const unsigned short* row, int group) {
    union { v16bf v; uint4 q[2]; } f;
    f.q[0] = *(const uint4*)(row + group * 8);
    f.q[1] = *(const uint4*)(row + 16 + group * 8);
    return f.v;
}

// B-matrix fragment (32x16 bf16, N=lane&15, K split per ISA layout like A), `row` points at the
// 32 contiguous K-values belonging to column N.
__device__ inline v16bf load_frag_b(const unsigned short* row, int group) {
    union { v16bf v; uint4 q[2]; } f;
    f.q[0] = *(const uint4*)(row + group * 8);
    f.q[1] = *(const uint4*)(row + 16 + group * 8);
    return f.v;
}

// ---------------------------------------------------------------------------
// Weight convert: fp32 (rows x cols) -> bf16 transposed (cols x rows)
// ---------------------------------------------------------------------------
__global__ void convert_wt_k(const float* __restrict__ src,
                             unsigned short* __restrict__ dst,
                             int rows, int cols) {
    int idx = blockIdx.x * 256 + threadIdx.x;
    if (idx >= rows * cols) return;
    int r = idx / cols, c = idx % cols;
    dst[(size_t)c * rows + r] = f2bf(src[idx]);
}

// ---------------------------------------------------------------------------
// Decorator stage 1: T[i][j] = bf16(tanh(y_i * W1[j] + b1[j]))
// ---------------------------------------------------------------------------
__global__ void build_T_k(const float* __restrict__ y, const float* __restrict__ W1,
                          const float* __restrict__ b1, unsigned short* __restrict__ T,
                          int nc) {
    int idx = blockIdx.x * 256 + threadIdx.x;
    int i = idx >> 8, j = idx & 255;
    float yv = (i < nc) ? y[i] : 0.0f;
    T[idx] = f2bf(tanhf(yv * W1[j] + b1[j]));
}

// ---------------------------------------------------------------------------
// Decorator stage 3: c = c_all + (w * ctx_mask) * yemb  (yemb already + b2)
// ---------------------------------------------------------------------------
__global__ void decorate_k(const float* __restrict__ cc, const float* __restrict__ ct,
                           const float* __restrict__ yemb, const float* __restrict__ ec,
                           const float* __restrict__ lk, float* __restrict__ cF,
                           unsigned short* __restrict__ cB, int nc) {
    int idx = blockIdx.x * 256 + threadIdx.x;
    int i = idx >> 8, j = idx & 255;
    float base = (i < nc) ? cc[(size_t)i * 256 + j] : ct[(size_t)(i - nc) * 256 + j];
    float val = base;
    if (i < nc) {
        float kappa = log1pf(__expf(lk[0]));   // softplus
        float e = ec[i];
        val += (e / (e + kappa)) * yemb[idx];
    }
    cF[idx] = val;
    cB[idx] = f2bf(val);
}

// ---------------------------------------------------------------------------
// Generic bf16 GEMM: C[MxN] = A[MxK] * Bt[NxK]^T + bias
// MODE: 0 = f32 out, 1 = bf16 out, 2 = bf16 transposed out, 3 = bf16 out + relu
// Block: 256 thr = 8 waves (4x2), tile 128x128; wave tile 32x64 -> 8 WMMA per
// 32-wide K step from 12 LDS fragment loads.
// ---------------------------------------------------------------------------
template <int MODE>
__global__ __launch_bounds__(256) void gemm_bf16_k(
    const unsigned short* __restrict__ A, const unsigned short* __restrict__ Bt,
    const float* __restrict__ bias, void* __restrict__ Cout,
    int M, int N, int K) {
    __shared__ unsigned short As[128][40];
    __shared__ unsigned short Bs[128][40];

    const int tid  = threadIdx.x;
    const int lane = tid & 31, wave = tid >> 5;
    const int wm = wave >> 1, wn = wave & 1;         // 4 x 2 wave grid
    const int group = lane >> 4, mrow = lane & 15;
    const int m0 = blockIdx.y * 128, n0 = blockIdx.x * 128;

    union Acc { v8f v; float f[8]; };
    Acc acc[2][4];
#pragma unroll
    for (int i = 0; i < 2; i++)
#pragma unroll
        for (int j = 0; j < 4; j++)
#pragma unroll
            for (int e = 0; e < 8; e++) acc[i][j].f[e] = 0.0f;

    const int ldrow = tid >> 1, ldc = (tid & 1) * 16;

    for (int k0 = 0; k0 < K; k0 += 32) {
        __syncthreads();
        const unsigned short* sa = A + (size_t)(m0 + ldrow) * K + k0 + ldc;
        const unsigned short* sb = Bt + (size_t)(n0 + ldrow) * K + k0 + ldc;
#if HAVE_ASYNC_LDS
        __builtin_amdgcn_global_load_async_to_lds_b128(GPTR(sa),     LPTR(&As[ldrow][ldc]),     0, 0);
        __builtin_amdgcn_global_load_async_to_lds_b128(GPTR(sa + 8), LPTR(&As[ldrow][ldc + 8]), 0, 0);
        __builtin_amdgcn_global_load_async_to_lds_b128(GPTR(sb),     LPTR(&Bs[ldrow][ldc]),     0, 0);
        __builtin_amdgcn_global_load_async_to_lds_b128(GPTR(sb + 8), LPTR(&Bs[ldrow][ldc + 8]), 0, 0);
#else
        *(uint4*)&As[ldrow][ldc]     = *(const uint4*)(sa);
        *(uint4*)&As[ldrow][ldc + 8] = *(const uint4*)(sa + 8);
        *(uint4*)&Bs[ldrow][ldc]     = *(const uint4*)(sb);
        *(uint4*)&Bs[ldrow][ldc + 8] = *(const uint4*)(sb + 8);
#endif
        if (k0 + 32 < K) {                 // warm L0/L2 for the next K step
            __builtin_prefetch(sa + 32, 0, 1);
            __builtin_prefetch(sb + 32, 0, 1);
        }
#if HAVE_ASYNC_LDS
        __builtin_amdgcn_s_wait_asynccnt(0);
#endif
        __syncthreads();

        v16bf af[2], bfr[4];
#pragma unroll
        for (int i = 0; i < 2; i++)
            af[i] = load_frag_a(&As[wm * 32 + i * 16 + mrow][0], group);
#pragma unroll
        for (int j = 0; j < 4; j++)
            bfr[j] = load_frag_b(&Bs[wn * 64 + j * 16 + mrow][0], group);
#pragma unroll
        for (int i = 0; i < 2; i++)
#pragma unroll
            for (int j = 0; j < 4; j++)
                acc[i][j].v = wmma_bf16(af[i], bfr[j], acc[i][j].v);
    }

#pragma unroll
    for (int j = 0; j < 4; j++) {
        const int col = n0 + wn * 64 + j * 16 + mrow;
        const float bv = bias[col];
#pragma unroll
        for (int i = 0; i < 2; i++)
#pragma unroll
            for (int e = 0; e < 8; e++) {
                int row = m0 + wm * 32 + i * 16 + e + 8 * group;  // C layout
                float v = acc[i][j].f[e] + bv;
                if constexpr (MODE == 3) v = v > 0.0f ? v : 0.0f;
                if constexpr (MODE == 0)
                    ((float*)Cout)[(size_t)row * N + col] = v;
                else if constexpr (MODE == 2)
                    ((unsigned short*)Cout)[(size_t)col * M + row] = f2bf(v);
                else
                    ((unsigned short*)Cout)[(size_t)row * N + col] = f2bf(v);
            }
    }
}

// ---------------------------------------------------------------------------
// Flash attention: per block (head, 128 queries); wave owns 16 queries.
// Q,K bf16 (n x 256) row-major; Vt bf16 (256 x n) dim-major; Out bf16 (n x 256).
// Streams keys in 64-tiles: S = Q Kt (4 wmma), online softmax, O += P V (4 wmma).
// Masking (target-target off-diagonal -> -inf) is only instantiated for key
// tiles overlapping the target region; the first NC/64 tiles use the clean path.
// ---------------------------------------------------------------------------
__global__ __launch_bounds__(256) void flash_attn_k(
    const unsigned short* __restrict__ Qm, const unsigned short* __restrict__ Km,
    const unsigned short* __restrict__ Vt, unsigned short* __restrict__ Om,
    int n, int nc) {
    __shared__ unsigned short Qs[128][40];
    __shared__ unsigned short Ks[64][40];      // keys x dims
    __shared__ unsigned short Vs[32][72];      // dims x keys
    __shared__ unsigned short Ps[8][16][72];   // per-wave probabilities

    const int tid = threadIdx.x, lane = tid & 31, wave = tid >> 5;
    const int group = lane >> 4, mrow = lane & 15;
    const int h  = blockIdx.y;
    const int q0 = blockIdx.x * 128;

    {   // stage Q tile (128 x 32 bf16)
        int row = tid >> 1, c16 = (tid & 1) * 16;
        const unsigned short* src = Qm + (size_t)(q0 + row) * 256 + h * 32 + c16;
#if HAVE_ASYNC_LDS
        __builtin_amdgcn_global_load_async_to_lds_b128(GPTR(src),     LPTR(&Qs[row][c16]),     0, 0);
        __builtin_amdgcn_global_load_async_to_lds_b128(GPTR(src + 8), LPTR(&Qs[row][c16 + 8]), 0, 0);
        __builtin_amdgcn_s_wait_asynccnt(0);
#else
        *(uint4*)&Qs[row][c16]     = *(const uint4*)(src);
        *(uint4*)&Qs[row][c16 + 8] = *(const uint4*)(src + 8);
#endif
    }
    __syncthreads();
    const v16bf aq = load_frag_a(&Qs[wave * 16 + mrow][0], group);

    union Acc { v8f v; float f[8]; };
    Acc o0, o1;
    float mr[8], lr[8];
    int qg[8];
    bool qT[8];
#pragma unroll
    for (int e = 0; e < 8; e++) {
        o0.f[e] = 0.f; o1.f[e] = 0.f; mr[e] = -1e30f; lr[e] = 0.f;
        qg[e] = q0 + wave * 16 + e + 8 * group;   // C-layout row
        qT[e] = qg[e] >= nc;
    }

    const float scale = 0.17677669529663687f;   // 1/sqrt(HD=32)
    const int krow = tid >> 2, kcol = (tid & 3) * 8;   // Ks staging: 64 x 32
    const int vrow = tid >> 3, vcol = (tid & 7) * 8;   // Vs staging: 32 x 64

    auto tile = [&](int kt, auto maskc) {
        constexpr bool MASK = decltype(maskc)::value;
        __syncthreads();
        const unsigned short* sk = Km + (size_t)(kt + krow) * 256 + h * 32 + kcol;
        const unsigned short* sv = Vt + (size_t)(h * 32 + vrow) * n + kt + vcol;
#if HAVE_ASYNC_LDS
        __builtin_amdgcn_global_load_async_to_lds_b128(GPTR(sk), LPTR(&Ks[krow][kcol]), 0, 0);
        __builtin_amdgcn_global_load_async_to_lds_b128(GPTR(sv), LPTR(&Vs[vrow][vcol]), 0, 0);
#else
        *(uint4*)&Ks[krow][kcol] = *(const uint4*)(sk);
        *(uint4*)&Vs[vrow][vcol] = *(const uint4*)(sv);
#endif
        if (kt + 64 < n) {                 // warm caches for next key tile
            __builtin_prefetch(sk + 64 * 256, 0, 1);
            __builtin_prefetch(sv + 64, 0, 1);
        }
#if HAVE_ASYNC_LDS
        __builtin_amdgcn_s_wait_asynccnt(0);
#endif
        __syncthreads();

        // scores: B = K^T -> column N is key, contiguous dims in Ks row
        Acc s[4];
        v8f z = {};
#pragma unroll
        for (int j = 0; j < 4; j++) {
            v16bf bk = load_frag_b(&Ks[j * 16 + mrow][0], group);
            s[j].v = wmma_bf16(aq, bk, z);
        }

#pragma unroll
        for (int e = 0; e < 8; e++) {
            float sv4[4];
#pragma unroll
            for (int j = 0; j < 4; j++) {
                sv4[j] = s[j].f[e] * scale;
                if constexpr (MASK) {
                    int kg = kt + j * 16 + mrow;
                    if (qT[e] && kg >= nc && qg[e] != kg) sv4[j] = -__builtin_inff();
                }
            }
            float rm = fmaxf(fmaxf(sv4[0], sv4[1]), fmaxf(sv4[2], sv4[3]));
            for (int msk = 8; msk >= 1; msk >>= 1)
                rm = fmaxf(rm, __shfl_xor(rm, msk, 16));
            float mnew  = fmaxf(mr[e], rm);
            float alpha = __expf(mr[e] - mnew);
            float rs = 0.0f;
            float pv[4];
#pragma unroll
            for (int j = 0; j < 4; j++) { pv[j] = __expf(sv4[j] - mnew); rs += pv[j]; }
            for (int msk = 8; msk >= 1; msk >>= 1)
                rs += __shfl_xor(rs, msk, 16);
            lr[e] = lr[e] * alpha + rs;
            mr[e] = mnew;
            o0.f[e] *= alpha;
            o1.f[e] *= alpha;
#pragma unroll
            for (int j = 0; j < 4; j++)
                Ps[wave][e + 8 * group][j * 16 + mrow] = f2bf(pv[j]);
        }
        // per-wave LDS ops are in-order: read P back in A layout (keys 0-31, 32-63)
        v16bf ap0 = load_frag_a(&Ps[wave][mrow][0],  group);
        v16bf ap1 = load_frag_a(&Ps[wave][mrow][32], group);
        v16bf bv00 = load_frag_b(&Vs[mrow][0],       group);  // keys 0-31,  dims 0-15
        v16bf bv01 = load_frag_b(&Vs[16 + mrow][0],  group);  // keys 0-31,  dims 16-31
        v16bf bv10 = load_frag_b(&Vs[mrow][32],      group);  // keys 32-63, dims 0-15
        v16bf bv11 = load_frag_b(&Vs[16 + mrow][32], group);  // keys 32-63, dims 16-31
        o0.v = wmma_bf16(ap0, bv00, o0.v);
        o0.v = wmma_bf16(ap1, bv10, o0.v);
        o1.v = wmma_bf16(ap0, bv01, o1.v);
        o1.v = wmma_bf16(ap1, bv11, o1.v);
    };

    for (int kt = 0; kt < n; kt += 64) {
        if (kt + 64 <= nc) tile(kt, BoolC<false>{});   // context-only: no mask
        else               tile(kt, BoolC<true>{});
    }

#pragma unroll
    for (int e = 0; e < 8; e++) {
        float inv = 1.0f / lr[e];
        Om[(size_t)qg[e] * 256 + h * 32 + mrow]      = f2bf(o0.f[e] * inv);
        Om[(size_t)qg[e] * 256 + h * 32 + 16 + mrow] = f2bf(o1.f[e] * inv);
    }
}

// ---------------------------------------------------------------------------
// Residual add + LayerNorm; one wave per row of 256
// ---------------------------------------------------------------------------
__global__ __launch_bounds__(256) void add_ln_k(
    const float* __restrict__ x, const float* resid,
    const float* __restrict__ g, const float* __restrict__ b,
    float* outF, unsigned short* __restrict__ outB) {
    int wave = threadIdx.x >> 5, lane = threadIdx.x & 31;
    int row = blockIdx.x * 8 + wave;
    float v[8];
    float s = 0.0f;
#pragma unroll
    for (int k = 0; k < 8; k++) {
        int col = lane + 32 * k;
        v[k] = x[(size_t)row * 256 + col] + resid[(size_t)row * 256 + col];
        s += v[k];
    }
    for (int m = 16; m >= 1; m >>= 1) s += __shfl_xor(s, m, 32);
    float mean = s * (1.0f / 256.0f);
    float q = 0.0f;
#pragma unroll
    for (int k = 0; k < 8; k++) { float d = v[k] - mean; q += d * d; }
    for (int m = 16; m >= 1; m >>= 1) q += __shfl_xor(q, m, 32);
    float rstd = rsqrtf(q * (1.0f / 256.0f) + 1e-5f);
#pragma unroll
    for (int k = 0; k < 8; k++) {
        int col = lane + 32 * k;
        float y = (v[k] - mean) * rstd * g[col] + b[col];
        outF[(size_t)row * 256 + col] = y;
        outB[(size_t)row * 256 + col] = f2bf(y);
    }
}

// ---------------------------------------------------------------------------
// Output head: out[t] = exp(dot(c[NC+t], W) + b) * exposure_target[t]
// ---------------------------------------------------------------------------
__global__ __launch_bounds__(256) void head_k(
    const float* __restrict__ c, const float* __restrict__ w,
    const float* __restrict__ b0, const float* __restrict__ expo,
    float* __restrict__ out, int nc) {
    int wave = threadIdx.x >> 5, lane = threadIdx.x & 31;
    int t = blockIdx.x * 8 + wave;
    const float* row = c + (size_t)(nc + t) * 256;
    float s = 0.0f;
#pragma unroll
    for (int k = 0; k < 8; k++) { int col = lane + 32 * k; s += row[col] * w[col]; }
    for (int m = 16; m >= 1; m >>= 1) s += __shfl_xor(s, m, 32);
    if (lane == 0) out[t] = __expf(s + b0[0]) * expo[t];
}

// ---------------------------------------------------------------------------
// Host orchestration
// ---------------------------------------------------------------------------
extern "C" void kernel_launch(void* const* d_in, const int* in_sizes, int n_in,
                              void* d_out, int out_size, void* d_ws, size_t ws_size,
                              hipStream_t stream) {
    (void)in_sizes; (void)n_in; (void)out_size; (void)ws_size;
    const float* c_ctx  = (const float*)d_in[0];
    const float* c_tgt  = (const float*)d_in[1];
    const float* y_ctx  = (const float*)d_in[2];
    const float* exp_c  = (const float*)d_in[3];
    const float* exp_t  = (const float*)d_in[4];
    const float* dec_W1 = (const float*)d_in[5];
    const float* dec_b1 = (const float*)d_in[6];
    const float* dec_W2 = (const float*)d_in[7];
    const float* dec_b2 = (const float*)d_in[8];
    const float* log_k  = (const float*)d_in[9];
    const float* WQ = (const float*)d_in[10]; const float* bQ = (const float*)d_in[11];
    const float* WK = (const float*)d_in[12]; const float* bK = (const float*)d_in[13];
    const float* WV = (const float*)d_in[14]; const float* bV = (const float*)d_in[15];
    const float* WO = (const float*)d_in[16]; const float* bO = (const float*)d_in[17];
    const float* ln1_g = (const float*)d_in[18]; const float* ln1_b = (const float*)d_in[19];
    const float* ln2_g = (const float*)d_in[20]; const float* ln2_b = (const float*)d_in[21];
    const float* fW1 = (const float*)d_in[22]; const float* fb1 = (const float*)d_in[23];
    const float* fW2 = (const float*)d_in[24]; const float* fb2 = (const float*)d_in[25];
    const float* out_W = (const float*)d_in[26]; const float* out_b = (const float*)d_in[27];

    constexpr int NC = 3072, NT = 1024, D = 256, H = 8, L = 2, FF = 1024;
    constexpr int n = NC + NT;

    char* p = (char*)d_ws;
    auto carve = [&](size_t bytes) -> void* {
        void* r = (void*)p;
        p += (bytes + 255) & ~(size_t)255;
        return r;
    };

    // bf16 transposed weights
    unsigned short* wDec = (unsigned short*)carve((size_t)D * D * 2);
    unsigned short *wq[L], *wk[L], *wv[L], *wo[L], *wf1[L], *wf2[L];
    for (int l = 0; l < L; l++) {
        wq[l]  = (unsigned short*)carve((size_t)D * D * 2);
        wk[l]  = (unsigned short*)carve((size_t)D * D * 2);
        wv[l]  = (unsigned short*)carve((size_t)D * D * 2);
        wo[l]  = (unsigned short*)carve((size_t)D * D * 2);
        wf1[l] = (unsigned short*)carve((size_t)D * FF * 2);  // (FF x D)
        wf2[l] = (unsigned short*)carve((size_t)FF * D * 2);  // (D x FF)
    }
    // activations
    float*          cF  = (float*)carve((size_t)n * D * 4);
    unsigned short* cB  = (unsigned short*)carve((size_t)n * D * 2);
    unsigned short* qB  = (unsigned short*)carve((size_t)n * D * 2);   // also T
    unsigned short* kB  = (unsigned short*)carve((size_t)n * D * 2);
    unsigned short* vtB = (unsigned short*)carve((size_t)D * n * 2);   // V transposed
    unsigned short* aB  = (unsigned short*)carve((size_t)n * D * 2);   // attn out
    unsigned short* f1B = (unsigned short*)carve((size_t)n * FF * 2);
    float*          gF  = (float*)carve((size_t)n * D * 4);            // yemb / o / f2

    const int gDD = (D * D + 255) / 256;
    const int gDF = (D * FF + 255) / 256;
    convert_wt_k<<<gDD, 256, 0, stream>>>(dec_W2, wDec, D, D);
    for (int l = 0; l < L; l++) {
        convert_wt_k<<<gDD, 256, 0, stream>>>(WQ + (size_t)l * D * D, wq[l], D, D);
        convert_wt_k<<<gDD, 256, 0, stream>>>(WK + (size_t)l * D * D, wk[l], D, D);
        convert_wt_k<<<gDD, 256, 0, stream>>>(WV + (size_t)l * D * D, wv[l], D, D);
        convert_wt_k<<<gDD, 256, 0, stream>>>(WO + (size_t)l * D * D, wo[l], D, D);
        convert_wt_k<<<gDF, 256, 0, stream>>>(fW1 + (size_t)l * D * FF, wf1[l], D, FF);
        convert_wt_k<<<gDF, 256, 0, stream>>>(fW2 + (size_t)l * FF * D, wf2[l], FF, D);
    }

    const dim3 gD(D / 128, n / 128);   // (2, 32)
    const dim3 gFF(FF / 128, n / 128); // (8, 32)

    // decorator
    build_T_k<<<n, 256, 0, stream>>>(y_ctx, dec_W1, dec_b1, qB, NC);
    gemm_bf16_k<0><<<gD, 256, 0, stream>>>(qB, wDec, dec_b2, gF, n, D, D);
    decorate_k<<<n, 256, 0, stream>>>(c_ctx, c_tgt, gF, exp_c, log_k, cF, cB, NC);

    for (int l = 0; l < L; l++) {
        gemm_bf16_k<1><<<gD, 256, 0, stream>>>(cB, wq[l], bQ + (size_t)l * D, qB, n, D, D);
        gemm_bf16_k<1><<<gD, 256, 0, stream>>>(cB, wk[l], bK + (size_t)l * D, kB, n, D, D);
        gemm_bf16_k<2><<<gD, 256, 0, stream>>>(cB, wv[l], bV + (size_t)l * D, vtB, n, D, D);

        flash_attn_k<<<dim3(n / 128, H), 256, 0, stream>>>(qB, kB, vtB, aB, n, NC);

        gemm_bf16_k<0><<<gD, 256, 0, stream>>>(aB, wo[l], bO + (size_t)l * D, gF, n, D, D);
        add_ln_k<<<n / 8, 256, 0, stream>>>(gF, cF, ln1_g + (size_t)l * D,
                                            ln1_b + (size_t)l * D, cF, cB);

        gemm_bf16_k<3><<<gFF, 256, 0, stream>>>(cB, wf1[l], fb1 + (size_t)l * FF, f1B, n, FF, D);
        gemm_bf16_k<0><<<gD, 256, 0, stream>>>(f1B, wf2[l], fb2 + (size_t)l * D, gF, n, D, FF);
        add_ln_k<<<n / 8, 256, 0, stream>>>(gF, cF, ln2_g + (size_t)l * D,
                                            ln2_b + (size_t)l * D, cF, cB);
    }

    head_k<<<NT / 8, 256, 0, stream>>>(cF, out_W, out_b, exp_t, (float*)d_out, NC);
}
